// MultiHeadAttention_51891794870426
// MI455X (gfx1250) — matrix operand order repaired and checked by reading, not compiled, not run on hardware
//
#include <hip/hip_runtime.h>
#include <hip/hip_bf16.h>

// ---------------- problem constants ----------------
constexpr int kS  = 2048;   // sequence length
constexpr int kD  = 1024;   // model dim
constexpr int kH  = 16;     // heads
constexpr int kHD = 64;     // head dim

typedef __attribute__((ext_vector_type(16))) __bf16 v16bf;
typedef __attribute__((ext_vector_type(8)))  float  v8f;

// ---------------------------------------------------------------------------
// f32 -> bf16 conversion (elementwise)
// ---------------------------------------------------------------------------
__global__ void cvt_f32_to_bf16(const float* __restrict__ in,
                                __bf16* __restrict__ out, int n) {
  int i = blockIdx.x * blockDim.x + threadIdx.x;
  if (i < n) out[i] = (__bf16)in[i];
}

// ---------------------------------------------------------------------------
// int32 mask -> u8 mask (16MB -> 4MB, becomes L2-resident across heads)
// ---------------------------------------------------------------------------
__global__ void pack_mask_u8(const int* __restrict__ in,
                             unsigned char* __restrict__ out, int n) {
  int i = blockIdx.x * blockDim.x + threadIdx.x;
  if (i < n) out[i] = (unsigned char)(in[i] != 0);
}

// ---------------------------------------------------------------------------
// bf16 WMMA GEMM:  Y[m,n] = sum_k A[m,k] * W[n,k]   (Y = A @ W^T)
// A: row-major [M,K] bf16.  W: row-major [N,K] bf16 (original torch layout).
// One wave per 16x64 output tile (A-fragment reused across 4 WMMAs).
// MODE 0: bf16 out, layout [H][M][64]   (Q/K per-head row-major)
// MODE 1: bf16 out, layout [H][64][M]   (V per-head transposed)
// MODE 2: f32  out, row-major [M,N]     (final projection -> d_out)
// ---------------------------------------------------------------------------
template <int MODE>
__global__ void gemm_wmma_bf16(const __bf16* __restrict__ A,
                               const __bf16* __restrict__ W,
                               void* __restrict__ outp,
                               int M, int N, int K) {
  const int lane  = threadIdx.x & 31;
  const int n0    = blockIdx.x * 64;
  const int m0    = blockIdx.y * 16;
  const int row   = lane & 15;                 // A-row / B-col / D-col for this lane
  const int abase = (lane < 16) ? 0 : 8;       // D row-chunk base (rows 8*(lane/16)+i)
  const int bbase = (lane < 16) ? 0 : 16;      // B per-lane K chunk base

  const __bf16* ap = A + (size_t)(m0 + row) * K + bbase;

  v8f acc[4] = {{}, {}, {}, {}};
  for (int k0 = 0; k0 < K; k0 += 32) {
    __builtin_prefetch(ap + k0 + 512, 0, 0);   // global_prefetch_b8
    v16bf a;
#pragma unroll
    for (int i = 0; i < 8; ++i) {              // A lane chunk {b..b+15}
      a[i]     = ap[k0 + i];
      a[i + 8] = ap[k0 + 8 + i];
    }
#pragma unroll
    for (int j = 0; j < 4; ++j) {
      const __bf16* bp = W + (size_t)(n0 + j * 16 + row) * K + bbase;
      v16bf b;
#pragma unroll
      for (int i = 0; i < 16; ++i) b[i] = bp[k0 + i];
      acc[j] = __builtin_amdgcn_wmma_f32_16x16x32_bf16(false, a, false, b,
                                                       (short)0, acc[j], false, false);
    }
  }

#pragma unroll
  for (int j = 0; j < 4; ++j) {
    if (MODE == 0) {                           // [H][M][64] bf16; n0 is 64-aligned
      __bf16* out = (__bf16*)outp;
      const int h = n0 >> 6;
      __bf16* dst = out + ((size_t)h * M + m0 + abase) * kHD + j * 16 + row;
#pragma unroll
      for (int i = 0; i < 8; ++i) dst[(size_t)i * kHD] = (__bf16)acc[j][i];
    } else if (MODE == 1) {                    // [H][64][M] bf16 (transposed per head)
      __bf16* out = (__bf16*)outp;
      const int h = n0 >> 6;
      __bf16* dst = out + ((size_t)h * kHD + j * 16 + row) * M + m0 + abase;
#pragma unroll
      for (int i = 0; i < 8; ++i) dst[i] = (__bf16)acc[j][i];
    } else {                                   // f32 row-major [M,N]
      float* out = (float*)outp;
      float* dst = out + (size_t)(m0 + abase) * N + n0 + j * 16 + row;
#pragma unroll
      for (int i = 0; i < 8; ++i) dst[(size_t)i * N] = acc[j][i];
    }
  }
}

// ---------------------------------------------------------------------------
// Fully fused attention block (per head, per 16-row block):
//   scores = (Q K^T)/8, masked_fill(mask, -1e-7)   -> kept in 128KB LDS stripe
//   out    = scores @ V  (pre-softmax, per reference) -> bf16 to concat
//   attn   = softmax(scores) from LDS               -> single HBM write
// 4 waves / block; each wave owns a 512-key range. Grid: (S/16, H).
// ---------------------------------------------------------------------------
__global__ void attn_fused(const __bf16* __restrict__ Qb,     // [H][S][64]
                           const __bf16* __restrict__ Kb,     // [H][S][64]
                           const __bf16* __restrict__ Vt,     // [H][64][S]
                           const unsigned char* __restrict__ mask8, // [S][S]
                           float* __restrict__ attn,          // [H][S][S]
                           __bf16* __restrict__ concat) {     // [S][D]
  extern __shared__ float sscores[];           // [16][kS]  = 128 KB
  __shared__ float opart[4][16][64];           // per-wave out partials (16 KB)
  __shared__ float rmaxp[16][8], rsump[16][8];
  __shared__ float rM[16], rSinv[16];

  float (*scores)[kS] = (float(*)[kS])sscores;

  const int tid   = threadIdx.x;               // 128 threads = 4 waves
  const int wave  = tid >> 5;
  const int lane  = tid & 31;
  const int m0    = blockIdx.x * 16;
  const int h     = blockIdx.y;
  const int row   = lane & 15;
  const int abase = (lane < 16) ? 0 : 8;
  const int bbase = (lane < 16) ? 0 : 16;

  const __bf16* Qh = Qb + (size_t)h * kS * kHD;
  const __bf16* Kh = Kb + (size_t)h * kS * kHD;
  const __bf16* Vh = Vt + (size_t)h * kHD * kS;

  // Q row-block A-fragments for the full d=64 reduction (two 32-wide steps)
  v16bf aq0, aq1;
  {
    const __bf16* qp = Qh + (size_t)(m0 + row) * kHD + bbase;
#pragma unroll
    for (int i = 0; i < 8; ++i) {
      aq0[i] = qp[i];       aq0[i + 8] = qp[8 + i];
      aq1[i] = qp[32 + i];  aq1[i + 8] = qp[40 + i];
    }
  }

  v8f oacc[4] = {{}, {}, {}, {}};              // this wave's out partial (16x64)

  const int tbeg = wave * (kS / 4);
  const int tend = tbeg + (kS / 4);
  for (int t0 = tbeg; t0 < tend; t0 += 32) {
    // ---- two 16x16 score tiles covering keys [t0, t0+32) ----
#pragma unroll
    for (int half = 0; half < 2; ++half) {
      const int tc = t0 + half * 16;
      const __bf16* kp = Kh + (size_t)(tc + row) * kHD + bbase;
      __builtin_prefetch(kp + 32 * kHD, 0, 0);
      v16bf b0, b1;
#pragma unroll
      for (int i = 0; i < 16; ++i) { b0[i] = kp[i]; b1[i] = kp[32 + i]; }
      v8f c = {};
      c = __builtin_amdgcn_wmma_f32_16x16x32_bf16(false, aq0, false, b0,
                                                  (short)0, c, false, false);
      c = __builtin_amdgcn_wmma_f32_16x16x32_bf16(false, aq1, false, b1,
                                                  (short)0, c, false, false);
      // scale + mask, park masked scores in the LDS stripe (pre-softmax)
#pragma unroll
      for (int i = 0; i < 8; ++i) {
        const int m = abase + i;
        const int t = tc + row;
        float s = c[i] * 0.125f;               // 1/sqrt(64)
        if (mask8[(size_t)(m0 + m) * kS + t]) s = -1e-7f;
        scores[m][t] = s;
      }
    }

    // ---- A-fragment straight from the LDS f32 stripe (cvt to bf16) ----
    v16bf as;
#pragma unroll
    for (int i = 0; i < 8; ++i) {
      as[i]     = (__bf16)scores[row][t0 + bbase + i];
      as[i + 8] = (__bf16)scores[row][t0 + bbase + 8 + i];
    }

    // ---- out += scores_tile @ V[t0:t0+32, :]  (4 n-tiles of 16) ----
#pragma unroll
    for (int j = 0; j < 4; ++j) {
      const __bf16* vp = Vh + (size_t)(j * 16 + row) * kS + t0 + bbase;
      v16bf bv;
#pragma unroll
      for (int i = 0; i < 16; ++i) bv[i] = vp[i];
      oacc[j] = __builtin_amdgcn_wmma_f32_16x16x32_bf16(false, as, false, bv,
                                                        (short)0, oacc[j], false, false);
    }
  }

  // ---- stash per-wave out partials ----
#pragma unroll
  for (int j = 0; j < 4; ++j)
#pragma unroll
    for (int i = 0; i < 8; ++i)
      opart[wave][abase + i][j * 16 + row] = oacc[j][i];
  __syncthreads();

  // ---- reduce 4 partials, write out tile (bf16) into concat [S][D] ----
#pragma unroll
  for (int e = 0; e < 8; ++e) {
    const int flat = e * 128 + tid;            // 0..1023 over the 16x64 tile
    const int m = flat >> 6, dc = flat & 63;
    float s = opart[0][m][dc] + opart[1][m][dc] + opart[2][m][dc] + opart[3][m][dc];
    concat[(size_t)(m0 + m) * kD + h * kHD + dc] = (__bf16)s;
  }

  // ---- softmax over the LDS stripe: 8 threads per row ----
  {
    const int r = tid >> 3, sub = tid & 7;
    float mx = -3.4e38f;
    for (int j = 0; j < kS / 8; ++j) mx = fmaxf(mx, scores[r][sub * (kS / 8) + j]);
    rmaxp[r][sub] = mx;
  }
  __syncthreads();
  if (tid < 16) {
    float mx = rmaxp[tid][0];
#pragma unroll
    for (int j = 1; j < 8; ++j) mx = fmaxf(mx, rmaxp[tid][j]);
    rM[tid] = mx;
  }
  __syncthreads();
  {
    const int r = tid >> 3, sub = tid & 7;
    const float mx = rM[r];
    float sum = 0.0f;
    for (int j = 0; j < kS / 8; ++j) {
      const int t = sub * (kS / 8) + j;
      float e = __expf(scores[r][t] - mx);
      scores[r][t] = e;                        // store exp in place
      sum += e;
    }
    rsump[r][sub] = sum;
  }
  __syncthreads();
  if (tid < 16) {
    float sum = rsump[tid][0];
#pragma unroll
    for (int j = 1; j < 8; ++j) sum += rsump[tid][j];
    rSinv[tid] = 1.0f / sum;
  }
  __syncthreads();

  // ---- single coalesced HBM write of normalized scores ----
  float* dst = attn + ((size_t)h * kS + m0) * kS;
  for (int c = 0; c < 256; ++c) {
    const int flat = c * 128 + tid;            // 0..32767 over the 16x2048 stripe
    const int m = flat >> 11, t = flat & (kS - 1);
    dst[(size_t)m * kS + t] = scores[m][t] * rSinv[m];
  }
}

// ---------------------------------------------------------------------------
// launch
// ---------------------------------------------------------------------------
extern "C" void kernel_launch(void* const* d_in, const int* in_sizes, int n_in,
                              void* d_out, int out_size, void* d_ws, size_t ws_size,
                              hipStream_t stream) {
  (void)in_sizes; (void)n_in; (void)out_size; (void)ws_size;

  const float* q    = (const float*)d_in[0];
  const float* k    = (const float*)d_in[1];
  const float* v    = (const float*)d_in[2];
  const int*   mask = (const int*)d_in[3];
  const float* Wq   = (const float*)d_in[4];
  const float* Wk   = (const float*)d_in[5];
  const float* Wv   = (const float*)d_in[6];
  const float* Wo   = (const float*)d_in[7];

  float* out_proj = (float*)d_out;                           // [S][D]
  float* attn     = (float*)d_out + (size_t)kS * kD;         // [H][S][S]

  // ---- workspace carve-up ----
  constexpr size_t nSD = (size_t)kS * kD;                    // 2M elems
  constexpr size_t nDD = (size_t)kD * kD;                    // 1M elems
  constexpr size_t nSS = (size_t)kS * kS;                    // 4M elems
  char* ws = (char*)d_ws;
  __bf16* Xq  = (__bf16*)(ws);                          // [S][D]
  __bf16* Xk  = Xq  + nSD;
  __bf16* Xv  = Xk  + nSD;
  __bf16* Wqb = Xv  + nSD;                              // [D][D]
  __bf16* Wkb = Wqb + nDD;
  __bf16* Wvb = Wkb + nDD;
  __bf16* Wob = Wvb + nDD;
  __bf16* Qb  = Wob + nDD;                              // [H][S][64]
  __bf16* Kb  = Qb  + nSD;                              // [H][S][64]
  __bf16* Vt  = Kb  + nSD;                              // [H][64][S]
  __bf16* cc  = Vt  + nSD;                              // concat [S][D]
  unsigned char* mask8 = (unsigned char*)(cc + nSD);    // [S][S] u8

  // 1) f32 -> bf16 converts + mask pack
  {
    const int tb = 256;
    cvt_f32_to_bf16<<<(int)(nSD + tb - 1) / tb, tb, 0, stream>>>(q,  Xq,  (int)nSD);
    cvt_f32_to_bf16<<<(int)(nSD + tb - 1) / tb, tb, 0, stream>>>(k,  Xk,  (int)nSD);
    cvt_f32_to_bf16<<<(int)(nSD + tb - 1) / tb, tb, 0, stream>>>(v,  Xv,  (int)nSD);
    cvt_f32_to_bf16<<<(int)(nDD + tb - 1) / tb, tb, 0, stream>>>(Wq, Wqb, (int)nDD);
    cvt_f32_to_bf16<<<(int)(nDD + tb - 1) / tb, tb, 0, stream>>>(Wk, Wkb, (int)nDD);
    cvt_f32_to_bf16<<<(int)(nDD + tb - 1) / tb, tb, 0, stream>>>(Wv, Wvb, (int)nDD);
    cvt_f32_to_bf16<<<(int)(nDD + tb - 1) / tb, tb, 0, stream>>>(Wo, Wob, (int)nDD);
    pack_mask_u8  <<<(int)(nSS + tb - 1) / tb, tb, 0, stream>>>(mask, mask8, (int)nSS);
  }

  // 2) projections: Q/K -> [H][S][64], V -> [H][64][S]
  dim3 gg(kD / 64, kS / 16), gb(32);
  gemm_wmma_bf16<0><<<gg, gb, 0, stream>>>(Xq, Wqb, Qb, kS, kD, kD);
  gemm_wmma_bf16<0><<<gg, gb, 0, stream>>>(Xk, Wkb, Kb, kS, kD, kD);
  gemm_wmma_bf16<1><<<gg, gb, 0, stream>>>(Xv, Wvb, Vt, kS, kD, kD);

  // 3) fused scores + mask + (pre-softmax scores)@V + softmax (LDS-resident)
  attn_fused<<<dim3(kS / 16, kH), dim3(128), 16 * kS * sizeof(float), stream>>>(
      Qb, Kb, Vt, mask8, attn, cc);

  // 4) output projection: concat @ Wo^T -> d_out (f32)
  gemm_wmma_bf16<2><<<gg, gb, 0, stream>>>(cc, Wob, out_proj, kS, kD, kD);
}
